// WindowCrossAttention_85736137162869
// MI455X (gfx1250) — matrix operand order repaired and checked by reading, not compiled, not run on hardware
//
#include <hip/hip_runtime.h>
#include <hip/hip_bf16.h>

typedef _Float16 half_t;
typedef __attribute__((ext_vector_type(16))) _Float16 v16h;
typedef __attribute__((ext_vector_type(8)))  _Float16 h8;
typedef __attribute__((ext_vector_type(8)))  float    v8f;

#define WSZ    16
#define NTOK   256
#define NHEADS 6
#define HD     32
#define DIM    192
#define BATCH  4
#define IMG    192
#define WDIV   12
#define NWIN   (BATCH * WDIV * WDIV)   // 576
#define INB_LD 208                     // padded LDS stride (halves) for 256x192 tile

// ---------------------------------------------------------------------------
// WMMA operand loaders, per CDNA5 ISA §7.12.2 (wave32 layouts).
// A (16xK f16): lane m (m=lane&15) holds row m. halves 0..7  -> K = base+0..7
//               halves 8..15 -> K = base+16..23, base = (lane<16 ? 0 : 8).
// B (Kx16 f16): lane n (n=lane&15) holds col n. halves 0..7  -> K = base+0..7
//               halves 8..15 -> K = base+16..23, base = (lane<16 ? 0 : 8).
// Both loaders = two 16B contiguous reads per lane (ds_load_b128 / global b128).
// ---------------------------------------------------------------------------
__device__ __forceinline__ v16h load_A16(const half_t* base, int ldh, int row0, int k0) {
  const int lane = threadIdx.x & 31;
  const int kb   = (lane < 16) ? 0 : 8;
  const half_t* p = base + (size_t)(row0 + (lane & 15)) * ldh + k0 + kb;
  h8 lo = *(const h8*)p;
  h8 hi = *(const h8*)(p + 16);
  v16h r;
#pragma unroll
  for (int i = 0; i < 8; ++i) { r[i] = lo[i]; r[i + 8] = hi[i]; }
  return r;
}

__device__ __forceinline__ v16h load_B16(const half_t* base, int ldh, int col0, int k0) {
  const int lane = threadIdx.x & 31;
  const int kb   = (lane < 16) ? 0 : 8;
  const half_t* p = base + (size_t)(col0 + (lane & 15)) * ldh + k0 + kb;
  h8 lo = *(const h8*)p;
  h8 hi = *(const h8*)(p + 16);
  v16h r;
#pragma unroll
  for (int i = 0; i < 8; ++i) { r[i] = lo[i]; r[i + 8] = hi[i]; }
  return r;
}

__device__ __forceinline__ v8f wmma16(v16h a, v16h b, v8f c) {
  return __builtin_amdgcn_wmma_f32_16x16x32_f16(false, a, false, b, (short)0, c, false, false);
}

// ---------------------------------------------------------------------------
// Async global->LDS copies (CDNA5 GLOBAL_LOAD_ASYNC_TO_LDS_*, tracked by
// ASYNCcnt). LDS dest address = low 32 bits of the generic address.
// Followed by s_wait_asynccnt before barrier.
// ---------------------------------------------------------------------------
__device__ __forceinline__ void async_ldg_b128(void* lds_ptr, const void* gptr) {
  unsigned lds = (unsigned)(unsigned long long)lds_ptr;
  asm volatile("global_load_async_to_lds_b128 %0, %1, off"
               :: "v"(lds), "v"(gptr) : "memory");
}
__device__ __forceinline__ void async_ldg_b32(void* lds_ptr, const void* gptr) {
  unsigned lds = (unsigned)(unsigned long long)lds_ptr;
  asm volatile("global_load_async_to_lds_b32 %0, %1, off"
               :: "v"(lds), "v"(gptr) : "memory");
}
__device__ __forceinline__ void wait_async0() {
  asm volatile("s_wait_asynccnt 0x0" ::: "memory");
}

// ===========================================================================
// Kernel 1: per-window Q / K / V projection (WMMA GEMMs).
//   Q -> qbuf [win][head][t][hd] f16, pre-scaled by 1/sqrt(hd)
//   K -> kbuf [win][head][t][hd] f16
//   V -> vbuf [win][head][hd][t] f16 (transposed: computed as Wv @ ref^T)
// Dynamic LDS: inb 256x208 f16 (106496B) + wb 384x192 f16 (147456B) = 253952B
// ===========================================================================
extern "C" __global__ void __launch_bounds__(256)
k_qkv(const float* __restrict__ x, const float* __restrict__ ref,
      const float* __restrict__ wq, const float* __restrict__ bq,
      const float* __restrict__ wkv, const float* __restrict__ bkv,
      half_t* __restrict__ qbuf, half_t* __restrict__ kbuf, half_t* __restrict__ vbuf)
{
  extern __shared__ half_t sm1[];
  half_t* inb = sm1;                   // [256][INB_LD]
  half_t* wb  = sm1 + NTOK * INB_LD;   // up to [384][192]

  const int tid = threadIdx.x, lane = tid & 31, wid = tid >> 5;
  const int win = blockIdx.x;
  const int b   = win / (WDIV * WDIV);
  const int rr  = win % (WDIV * WDIV);
  const int h0  = (rr / WDIV) * WSZ, w0 = (rr % WDIV) * WSZ;
  const int hi8 = (lane < 16) ? 0 : 8;

  // ---------------- phase A: Q = x_win @ Wq^T + bq ----------------
  for (int it = 0; it < 192; ++it) {
    int idx = it * 256 + tid;
    int c = idx >> 8, t = idx & 255;
    int gh = h0 + (t >> 4), gw = w0 + (t & 15);
    inb[t * INB_LD + c] = (half_t)x[(((size_t)b * DIM + c) * IMG + gh) * IMG + gw];
  }
  for (int it = 0; it < 144; ++it) {
    int idx = it * 256 + tid;
    wb[idx] = (half_t)wq[idx];
  }
  __syncthreads();

  const float scale = 0.17677669529663687f;  // 1/sqrt(32)
  for (int nt = 0; nt < 12; ++nt) {
    v16h Bt[6];
#pragma unroll
    for (int kk = 0; kk < 6; ++kk) Bt[kk] = load_B16(wb, DIM, nt * 16, kk * 32);
    const float bias = bq[nt * 16 + (lane & 15)];
    const int head = nt >> 1, hdo = (nt & 1) * 16;
    half_t* qp = qbuf + ((size_t)win * NHEADS + head) * NTOK * HD;
#pragma unroll
    for (int j = 0; j < 2; ++j) {
      const int mt = wid * 2 + j;
      v8f acc;
#pragma unroll
      for (int v = 0; v < 8; ++v) acc[v] = bias;
#pragma unroll
      for (int kk = 0; kk < 6; ++kk) {
        v16h At = load_A16(inb, INB_LD, mt * 16, kk * 32);
        acc = wmma16(At, Bt[kk], acc);
      }
#pragma unroll
      for (int v = 0; v < 8; ++v) {
        int t = mt * 16 + v + hi8;
        qp[t * HD + hdo + (lane & 15)] = (half_t)(acc[v] * scale);
      }
    }
  }
  __syncthreads();

  // ---------------- phase B: K,V from ref_win @ Wkv^T + bkv ----------------
  for (int it = 0; it < 192; ++it) {
    int idx = it * 256 + tid;
    int c = idx >> 8, t = idx & 255;
    int gh = h0 + (t >> 4), gw = w0 + (t & 15);
    inb[t * INB_LD + c] = (half_t)ref[(((size_t)b * DIM + c) * IMG + gh) * IMG + gw];
  }
  for (int it = 0; it < 288; ++it) {
    int idx = it * 256 + tid;
    wb[idx] = (half_t)wkv[idx];
  }
  __syncthreads();

  // K (rows 0..191 of wkv), normal orientation -> [t][hd]
  for (int nt = 0; nt < 12; ++nt) {
    v16h Bt[6];
#pragma unroll
    for (int kk = 0; kk < 6; ++kk) Bt[kk] = load_B16(wb, DIM, nt * 16, kk * 32);
    const float bias = bkv[nt * 16 + (lane & 15)];
    const int head = nt >> 1, hdo = (nt & 1) * 16;
    half_t* kp = kbuf + ((size_t)win * NHEADS + head) * NTOK * HD;
#pragma unroll
    for (int j = 0; j < 2; ++j) {
      const int mt = wid * 2 + j;
      v8f acc;
#pragma unroll
      for (int v = 0; v < 8; ++v) acc[v] = bias;
#pragma unroll
      for (int kk = 0; kk < 6; ++kk) {
        v16h At = load_A16(inb, INB_LD, mt * 16, kk * 32);
        acc = wmma16(At, Bt[kk], acc);
      }
#pragma unroll
      for (int v = 0; v < 8; ++v) {
        int t = mt * 16 + v + hi8;
        kp[t * HD + hdo + (lane & 15)] = (half_t)acc[v];
      }
    }
  }

  // V (rows 192..383 of wkv), swapped: C = Wv(oc x c) @ ref^T(c x t) -> vT[hd][t]
#pragma unroll
  for (int j = 0; j < 2; ++j) {
    const int nt = wid * 2 + j;   // token tile
    v16h Bt[6];
#pragma unroll
    for (int kk = 0; kk < 6; ++kk) Bt[kk] = load_B16(inb, INB_LD, nt * 16, kk * 32);
    for (int mt = 0; mt < 12; ++mt) {   // oc tile
      v8f acc;
#pragma unroll
      for (int v = 0; v < 8; ++v) acc[v] = bkv[DIM + mt * 16 + v + hi8];
#pragma unroll
      for (int kk = 0; kk < 6; ++kk) {
        v16h At = load_A16(wb, DIM, DIM + mt * 16, kk * 32);
        acc = wmma16(At, Bt[kk], acc);
      }
      const int head = mt >> 1, hdo = (mt & 1) * 16;
      half_t* vp = vbuf + ((size_t)win * NHEADS + head) * HD * NTOK;
#pragma unroll
      for (int v = 0; v < 8; ++v) {
        int hd = hdo + v + hi8;
        vp[hd * NTOK + nt * 16 + (lane & 15)] = (half_t)acc[v];
      }
    }
  }
}

// ===========================================================================
// Kernel 2: attention per (window, head). 4 waves; wave owns 4 tiles of 16 query rows.
//   q/k/vT + RPE table staged with GLOBAL_LOAD_ASYNC_TO_LDS (ASYNCcnt path:
//   pure byte copies, no VGPR round-trip), then
//   S = Q K^T (+RPE bias in accumulator), softmax in regs (16-lane shfl),
//   O = P V via LDS-bounced A operand. Output obuf [win][t][DIM] f16.
// ===========================================================================
extern "C" __global__ void __launch_bounds__(128)
k_attn(const half_t* __restrict__ qbuf, const half_t* __restrict__ kbuf,
       const half_t* __restrict__ vbuf, const float* __restrict__ rpb,
       half_t* __restrict__ obuf)
{
  __shared__ __align__(16) half_t sq[NTOK * HD];
  __shared__ __align__(16) half_t sk[NTOK * HD];
  __shared__ __align__(16) half_t sv[HD * NTOK];
  __shared__ __align__(16) half_t sp[4][16 * 32];
  __shared__ float srpb[961];

  const int tid = threadIdx.x, lane = tid & 31, wid = tid >> 5;
  const int head = blockIdx.x % NHEADS;
  const int win  = blockIdx.x / NHEADS;
  const size_t off = ((size_t)win * NHEADS + head) * NTOK * HD;

  // ---- async staging: 3 x 16KB f16 copies + 3844B f32 RPE copy ----
  const uint4* gq = (const uint4*)(qbuf + off);
  const uint4* gk = (const uint4*)(kbuf + off);
  const uint4* gv = (const uint4*)(vbuf + off);
  for (int i = tid; i < (NTOK * HD) / 8; i += 128) {
    async_ldg_b128(&((uint4*)sq)[i], &gq[i]);
    async_ldg_b128(&((uint4*)sk)[i], &gk[i]);
    async_ldg_b128(&((uint4*)sv)[i], &gv[i]);
  }
  for (int i = tid; i < 961; i += 128)
    async_ldg_b32(&srpb[i], &rpb[head * 961 + i]);
  wait_async0();
  __syncthreads();

  const int hi8 = (lane < 16) ? 0 : 8;
  half_t* pw = sp[wid];

  for (int rt = wid; rt < 16; rt += 4) {
    v16h Aq = load_A16(sq, HD, rt * 16, 0);
    v8f S[16];
#pragma unroll
    for (int ct = 0; ct < 16; ++ct) {
      const int relh = ct - rt + 15;           // uniform per tile
      v8f acc;
#pragma unroll
      for (int v = 0; v < 8; ++v)              // relw = (lane&15) - (v+hi8) + 15
        acc[v] = srpb[relh * 31 + (lane & 15) - v - hi8 + 15];
      v16h Bk = load_B16(sk, HD, ct * 16, 0);
      S[ct] = wmma16(Aq, Bk, acc);             // hd=32 => single K-step
    }

    // ----- softmax across 256 keys: per-lane partial + 16-lane xor reduce -----
#pragma unroll
    for (int v = 0; v < 8; ++v) {
      float m = -1e30f;
#pragma unroll
      for (int ct = 0; ct < 16; ++ct) m = fmaxf(m, S[ct][v]);
#pragma unroll
      for (int msk = 1; msk < 16; msk <<= 1) m = fmaxf(m, __shfl_xor(m, msk));
      float s = 0.f;
#pragma unroll
      for (int ct = 0; ct < 16; ++ct) { float e = __expf(S[ct][v] - m); S[ct][v] = e; s += e; }
#pragma unroll
      for (int msk = 1; msk < 16; msk <<= 1) s += __shfl_xor(s, msk);
      const float inv = 1.0f / s;
#pragma unroll
      for (int ct = 0; ct < 16; ++ct) S[ct][v] *= inv;
    }

    // ----- O = P @ V : 8 chunks of 32 keys -----
    v8f O0, O1;
#pragma unroll
    for (int v = 0; v < 8; ++v) { O0[v] = 0.f; O1[v] = 0.f; }
    for (int kc = 0; kc < 8; ++kc) {
#pragma unroll
      for (int hc = 0; hc < 2; ++hc) {
        const int ct = kc * 2 + hc;
#pragma unroll
        for (int v = 0; v < 8; ++v)
          pw[(v + hi8) * 32 + hc * 16 + (lane & 15)] = (half_t)S[ct][v];
      }
      asm volatile("s_wait_dscnt 0x0" ::: "memory");
      v16h Ap  = load_A16(pw, 32, 0, 0);
      v16h Bv0 = load_B16(sv, NTOK, 0,  kc * 32);
      v16h Bv1 = load_B16(sv, NTOK, 16, kc * 32);
      O0 = wmma16(Ap, Bv0, O0);
      O1 = wmma16(Ap, Bv1, O1);
    }

    half_t* op = obuf + (size_t)win * NTOK * DIM + head * HD;
#pragma unroll
    for (int v = 0; v < 8; ++v) {
      const int t = rt * 16 + v + hi8;
      op[(size_t)t * DIM + (lane & 15)]      = (half_t)O0[v];
      op[(size_t)t * DIM + 16 + (lane & 15)] = (half_t)O1[v];
    }
  }
}

// ===========================================================================
// Kernel 3: out projection, swapped orientation (C tile = oc x t) so lanes map
// to consecutive w-pixels -> coalesced f32 stores into [B,C,H,W].
// A = W_out (LDS f16), B = O^T read directly from global obuf [t][DIM].
// Dynamic LDS: 192*192 f16 + 192 f32 = 74496 B
// ===========================================================================
extern "C" __global__ void __launch_bounds__(256)
k_out(const half_t* __restrict__ obuf, const float* __restrict__ wout,
      const float* __restrict__ bout, float* __restrict__ out)
{
  extern __shared__ half_t sm3[];
  half_t* wb = sm3;                          // [192][192]
  float*  sb = (float*)(sm3 + DIM * DIM);    // [192]

  const int tid = threadIdx.x, lane = tid & 31, wid = tid >> 5;
  const int win = blockIdx.x;
  const int b   = win / (WDIV * WDIV);
  const int rr  = win % (WDIV * WDIV);
  const int h0  = (rr / WDIV) * WSZ, w0 = (rr % WDIV) * WSZ;
  const int hi8 = (lane < 16) ? 0 : 8;

  for (int i = tid; i < DIM * DIM; i += 256) wb[i] = (half_t)wout[i];
  for (int i = tid; i < DIM; i += 256) sb[i] = bout[i];
  __syncthreads();

  const half_t* ob = obuf + (size_t)win * NTOK * DIM;
#pragma unroll
  for (int j = 0; j < 2; ++j) {
    const int nt = wid * 2 + j;              // token tile -> fixed row gh
    v16h Bt[6];
#pragma unroll
    for (int kk = 0; kk < 6; ++kk) Bt[kk] = load_B16(ob, DIM, nt * 16, kk * 32);
    const int gh = h0 + nt;
    for (int mt = 0; mt < 12; ++mt) {        // oc tile
      v8f acc;
#pragma unroll
      for (int v = 0; v < 8; ++v) acc[v] = sb[mt * 16 + v + hi8];
#pragma unroll
      for (int kk = 0; kk < 6; ++kk) {
        v16h At = load_A16(wb, DIM, mt * 16, kk * 32);
        acc = wmma16(At, Bt[kk], acc);
      }
#pragma unroll
      for (int v = 0; v < 8; ++v) {
        const int oc = mt * 16 + v + hi8;
        out[(((size_t)b * DIM + oc) * IMG + gh) * IMG + w0 + (lane & 15)] = acc[v];
      }
    }
  }
}

// ===========================================================================
extern "C" void kernel_launch(void* const* d_in, const int* in_sizes, int n_in,
                              void* d_out, int out_size, void* d_ws, size_t ws_size,
                              hipStream_t stream) {
  (void)in_sizes; (void)n_in; (void)out_size; (void)ws_size;
  const float* x    = (const float*)d_in[0];
  const float* ref  = (const float*)d_in[1];
  const float* wq   = (const float*)d_in[2];
  const float* bq   = (const float*)d_in[3];
  const float* wkv  = (const float*)d_in[4];
  const float* bkv  = (const float*)d_in[5];
  const float* wout = (const float*)d_in[6];
  const float* bout = (const float*)d_in[7];
  const float* rpb  = (const float*)d_in[8];
  float* outp = (float*)d_out;

  // f16 scratch partition (halves): q | k | v | o, each 576*6*256*32 = 28,311,552
  const size_t QSZ = (size_t)NWIN * NHEADS * NTOK * HD;
  half_t* qbuf = (half_t*)d_ws;
  half_t* kbuf = qbuf + QSZ;
  half_t* vbuf = kbuf + QSZ;
  half_t* obuf = vbuf + QSZ;   // NWIN*NTOK*DIM == QSZ

  const int lds1 = NTOK * INB_LD * 2 + 2 * DIM * DIM * 2;   // 253,952 B
  const int lds3 = DIM * DIM * 2 + DIM * 4;                 //  74,496 B
  hipFuncSetAttribute((const void*)k_qkv, hipFuncAttributeMaxDynamicSharedMemorySize, lds1);
  hipFuncSetAttribute((const void*)k_out, hipFuncAttributeMaxDynamicSharedMemorySize, lds3);

  k_qkv<<<NWIN, 256, lds1, stream>>>(x, ref, wq, bq, wkv, bkv, qbuf, kbuf, vbuf);
  k_attn<<<NWIN * NHEADS, 128, 0, stream>>>(qbuf, kbuf, vbuf, rpb, obuf);
  k_out<<<NWIN, 256, lds3, stream>>>(obuf, wout, bout, outp);
}
